// SimGSMN_89361089561125
// MI455X (gfx1250) — compile-verified, matrix operand order checked by reading, not run on hardware
//
#include <hip/hip_runtime.h>
#include <hip/hip_bf16.h>

typedef __attribute__((ext_vector_type(16))) _Float16 v16h;
typedef __attribute__((ext_vector_type(8)))  float    v8f;

#define Bn    2048
#define Nn    36
#define EMBD  1024
#define SIMD  16
#define BLKD  64     // EMBD / SIMD
#define HIDD  32
#define KKn   8

#define NPAD    48      // rows padded to 3x16 WMMA tiles
#define ASTRIDE 1032    // halfs per LDS row (1024 + 8 pad for bank spread)

#define SMEM_BYTES (NPAD*ASTRIDE*2 /*As f16*/ \
                  + Nn*SIMD*4 /*simv*/ + Nn*SIMD*4 /*qn2*/ + NPAD*4 /*rinv*/ \
                  + Nn*Nn*4 /*G*/ + KKn*Nn*4 /*S*/ \
                  + Nn*HIDD*4 /*hid*/ + Nn*HIDD*4 /*h1*/ \
                  + HIDD*HIDD*4 /*w1*/ + HIDD*4 /*b1*/ + HIDD*4 /*w2*/ + 4 /*b2*/ \
                  + Nn*4 /*sv*/ + 12 /*align slack*/)

__global__ __launch_bounds__(256)
void simgsmn_fused(const float* __restrict__ inp1, const float* __restrict__ inp2,
                   const float* __restrict__ gk_mean, const float* __restrict__ gk_prec,
                   const float* __restrict__ gcn_w,
                   const float* __restrict__ out1_v, const float* __restrict__ out1_g,
                   const float* __restrict__ out1_b,
                   const float* __restrict__ out2_v, const float* __restrict__ out2_g,
                   const float* __restrict__ out2_b,
                   float* __restrict__ out)
{
    extern __shared__ __align__(16) char smem[];
    _Float16* As  = (_Float16*)smem;                          // NPAD x ASTRIDE f16
    float* simv = (float*)(smem + NPAD*ASTRIDE*2);            // Nn x SIMD
    float* qn2  = simv + Nn*SIMD;                             // Nn x SIMD
    float* rinv = qn2  + Nn*SIMD;                             // NPAD
    float* G    = rinv + NPAD;                                // Nn x Nn
    float* S    = G    + Nn*Nn;                               // KKn x Nn
    float* hid  = S    + KKn*Nn;                              // Nn x HIDD
    float* h1   = hid  + Nn*HIDD;                             // Nn x HIDD
    float* w1   = h1   + Nn*HIDD;                             // HIDD x HIDD
    float* b1s  = w1   + HIDD*HIDD;                           // HIDD
    float* w2s  = b1s  + HIDD;                                // HIDD
    float* b2s  = w2s  + HIDD;                                // 1
    float* sv   = b2s  + 1;                                   // Nn

    const int tid  = threadIdx.x;
    const int b    = blockIdx.x;
    const size_t base = (size_t)b * Nn * EMBD;

    // ---- Phase 0: zero the pad rows (36..47) so WMMA tiles read zeros ----
    for (int idx = tid; idx < (NPAD - Nn) * ASTRIDE; idx += 256)
        As[Nn*ASTRIDE + idx] = (_Float16)0.0f;

    // ---- Phase 1: single streaming pass over inp1/inp2 -------------------
    // per (row i, block d): block norms + dot -> sim_vec; stage inp1 as f16
    for (int job = tid; job < Nn * SIMD; job += 256) {
        const int i = job >> 4, d = job & 15;
        const float4* q4 = (const float4*)(inp1 + base + (size_t)i*EMBD + d*BLKD);
        const float4* c4 = (const float4*)(inp2 + base + (size_t)i*EMBD + d*BLKD);
        _Float16* ad = As + i*ASTRIDE + d*BLKD;
        float sq = 0.f, sc = 0.f, dot = 0.f;
        #pragma unroll
        for (int e = 0; e < BLKD/4; ++e) {
            float4 q = q4[e], c = c4[e];
            sq  += q.x*q.x + q.y*q.y + q.z*q.z + q.w*q.w;
            sc  += c.x*c.x + c.y*c.y + c.z*c.z + c.w*c.w;
            dot += q.x*c.x + q.y*c.y + q.z*c.z + q.w*c.w;
            ad[4*e+0] = (_Float16)q.x; ad[4*e+1] = (_Float16)q.y;
            ad[4*e+2] = (_Float16)q.z; ad[4*e+3] = (_Float16)q.w;
        }
        qn2[job]  = sq;
        simv[job] = dot / ((sqrtf(sq) + 1e-8f) * (sqrtf(sc) + 1e-8f));
    }
    __syncthreads();

    // full-row inverse norms for xn = x / (||x|| + 1e-8)
    if (tid < Nn) {
        float s = 0.f;
        #pragma unroll
        for (int d = 0; d < SIMD; ++d) s += qn2[tid*SIMD + d];
        rinv[tid] = 1.0f / (sqrtf(s) + 1e-8f);
    }
    __syncthreads();

    // ---- Phase 2: 36x36 gram = A @ A^T via WMMA f16->f32 -----------------
    // 9 tiles of 16x16, one tile per wave (wave 0 takes tile 8 too).
    const int wave   = tid >> 5;
    const int lane   = tid & 31;
    const int laneLo = lane & 15;
    const int laneHi = lane >> 4;
    for (int t = wave; t < 9; t += 8) {
        const int r0 = (t / 3) * 16, c0 = (t % 3) * 16;
        v8f acc = {};
        const unsigned* arow = (const unsigned*)(As + (r0 + laneLo) * ASTRIDE);
        const unsigned* brow = (const unsigned*)(As + (c0 + laneLo) * ASTRIDE);
        for (int kk = 0; kk < EMBD; kk += 32) {
            union { v16h h; unsigned u[8]; } af, bf;
            #pragma unroll
            for (int u = 0; u < 8; ++u) {
                // 16-bit A-matrix 16x32 layout: dword u holds K pair
                const int Kb = ((u & 3) << 1) + ((u >= 4) ? 16 : 0) + (laneHi << 3);
                af.u[u] = arow[(kk + Kb) >> 1];
                bf.u[u] = brow[(kk + Kb) >> 1];   // B = A^T: symmetric gather
            }
            acc = __builtin_amdgcn_wmma_f32_16x16x32_f16(
                      false, af.h, false, bf.h, (short)0, acc, false, false);
        }
        #pragma unroll
        for (int r = 0; r < 8; ++r) {
            const int i = r0 + r + laneHi * 8;
            const int j = c0 + laneLo;
            if (i < Nn && j < Nn) G[i*Nn + j] = acc[r] * rinv[i] * rinv[j];
        }
    }
    __syncthreads();

    // ---- Phase 3: Gaussian kernels -> S[k,i] = sum_j w / (sum_j w + eps) -
    for (int job = tid; job < Nn * KKn; job += 256) {
        const int i = job >> 3, k = job & 7;
        const float mean = gk_mean[k], prec = gk_prec[k];
        float sum = 0.f;
        for (int j = 0; j < Nn; ++j) {
            float g  = G[i*Nn + j];
            float d2 = 2.0f - 2.0f * g; d2 = d2 > 0.f ? d2 : 0.f;
            float dd = sqrtf(d2 + 1e-12f);
            float u  = dd - mean;
            sum += __expf(-0.5f * prec * u * u);
        }
        S[k*Nn + i] = sum / (sum + 1e-8f);
    }
    __syncthreads();

    // ---- Phase 4: hidden[i,h] = sum_k S[k,i] * (sim[i,:] @ gcn_w[k]) -----
    for (int job = tid; job < Nn * HIDD; job += 256) {
        const int i = job >> 5, h = job & 31;
        float acc = 0.f;
        #pragma unroll
        for (int k = 0; k < KKn; ++k) {
            const float s = S[k*Nn + i];
            float p = 0.f;
            #pragma unroll
            for (int d = 0; d < SIMD; ++d)
                p += simv[i*SIMD + d] * gcn_w[(k*SIMD + d)*HIDD + h];
            acc += s * p;
        }
        hid[job] = acc;
    }

    // ---- Phase 5a: materialize weight-normed MLP weights -----------------
    if (tid < HIDD) {
        float nv = 0.f;
        #pragma unroll
        for (int c = 0; c < HIDD; ++c) { float v = out1_v[tid*HIDD + c]; nv += v*v; }
        const float sc = out1_g[tid] / (sqrtf(nv) + 1e-12f);
        #pragma unroll
        for (int c = 0; c < HIDD; ++c) w1[tid*HIDD + c] = out1_v[tid*HIDD + c] * sc;
        b1s[tid] = out1_b[tid];
    } else if (tid == HIDD) {
        float nv = 0.f;
        for (int c = 0; c < HIDD; ++c) { float v = out2_v[c]; nv += v*v; }
        const float sc = out2_g[0] / (sqrtf(nv) + 1e-12f);
        for (int c = 0; c < HIDD; ++c) w2s[c] = out2_v[c] * sc;
        b2s[0] = out2_b[0];
    }
    __syncthreads();

    // ---- Phase 5b: tanh(h @ w1^T + b1), then w2 dot, then mean over i ----
    for (int job = tid; job < Nn * HIDD; job += 256) {
        const int i = job >> 5, o = job & 31;
        float acc = b1s[o];
        #pragma unroll
        for (int c = 0; c < HIDD; ++c) acc += hid[i*HIDD + c] * w1[o*HIDD + c];
        h1[job] = tanhf(acc);
    }
    __syncthreads();
    if (tid < Nn) {
        float acc = b2s[0];
        #pragma unroll
        for (int c = 0; c < HIDD; ++c) acc += h1[tid*HIDD + c] * w2s[c];
        sv[tid] = acc;
    }
    __syncthreads();
    if (tid == 0) {
        float acc = 0.f;
        for (int i = 0; i < Nn; ++i) acc += sv[i];
        out[b] = acc * (1.0f / (float)Nn);
    }
}

extern "C" void kernel_launch(void* const* d_in, const int* in_sizes, int n_in,
                              void* d_out, int out_size, void* d_ws, size_t ws_size,
                              hipStream_t stream) {
    const float* inp1    = (const float*)d_in[0];
    const float* inp2    = (const float*)d_in[1];
    const float* gk_mean = (const float*)d_in[2];
    const float* gk_prec = (const float*)d_in[3];
    const float* gcn_w   = (const float*)d_in[4];
    const float* out1_v  = (const float*)d_in[5];
    const float* out1_g  = (const float*)d_in[6];
    const float* out1_b  = (const float*)d_in[7];
    const float* out2_v  = (const float*)d_in[8];
    const float* out2_g  = (const float*)d_in[9];
    const float* out2_b  = (const float*)d_in[10];
    float* out = (float*)d_out;

    simgsmn_fused<<<Bn, 256, SMEM_BYTES, stream>>>(
        inp1, inp2, gk_mean, gk_prec, gcn_w,
        out1_v, out1_g, out1_b, out2_v, out2_g, out2_b, out);
}
